// NormalizingFlowModel_12232066859404
// MI455X (gfx1250) — matrix operand order repaired and checked by reading, not compiled, not run on hardware
//
#include <hip/hip_runtime.h>
#include <math.h>

typedef __attribute__((ext_vector_type(16))) _Float16 v16h;
typedef __attribute__((ext_vector_type(8)))  float    v8f;

#define TT 5
#define DD 8
#define HH 128
#define KKB 8
#define PPB 23
#define BOUNDF 3.0f
#define MINBIN 0.001f

// ---------------- weight prep: fold masks, cast f16, swizzle to WMMA B-fragment layout ----------
// B fragment (32x16, 16-bit): lane = (n&15) + ((kl>>3)&1)*16; half j = ((kl>>4)&1)*8 + (kl&7), kl = k&31.

__global__ void prep_w1(const float* __restrict__ W1, _Float16* __restrict__ w1b) {
    int idx = blockIdx.x * blockDim.x + threadIdx.x;
    const int per_t = 96 * 128;
    if (idx >= TT * per_t) return;
    int t = idx / per_t, r = idx % per_t;
    int k = r >> 7, n = r & 127;               // k in 0..95 (K padded 72->96), n in 0..127
    float v = 0.f;
    if (k < 72) {
        int m_in = (k < DD) ? (k + 1) : 0;     // first 8 cols = xin, rest = ctx (mask 1)
        int m_h  = n % 7 + 1;
        if (m_h >= m_in) v = W1[(t * HH + n) * 72 + k];
    }
    int seg = (n >> 4) * 3 + (k >> 5);
    int kl = k & 31;
    int lane = (n & 15) + ((kl >> 3) & 1) * 16;
    w1b[t * (24 * 512) + (seg * 32 + lane) * 16 + (((kl >> 4) & 1) * 8 + (kl & 7))] = (_Float16)v;
}

__global__ void prep_w2(const float* __restrict__ W2, _Float16* __restrict__ w2b) {
    int idx = blockIdx.x * blockDim.x + threadIdx.x;
    const int per_t = 128 * 128;
    if (idx >= TT * per_t) return;
    int t = idx / per_t, r = idx % per_t;
    int k = r >> 7, n = r & 127;
    float v = 0.f;
    int m_hn = n % 7 + 1, m_hk = k % 7 + 1;
    if (m_hn >= m_hk) v = W2[(t * HH + n) * HH + k];
    int seg = (n >> 4) * 4 + (k >> 5);
    int kl = k & 31;
    int lane = (n & 15) + ((kl >> 3) & 1) * 16;
    w2b[t * (32 * 512) + (seg * 32 + lane) * 16 + (((kl >> 4) & 1) * 8 + (kl & 7))] = (_Float16)v;
}

__global__ void prep_w3(const float* __restrict__ W3, _Float16* __restrict__ w3b) {
    int idx = blockIdx.x * blockDim.x + threadIdx.x;
    const int per_t = 128 * 192;
    if (idx >= TT * per_t) return;
    int t = idx / per_t, r = idx % per_t;
    int k = r / 192, n = r % 192;              // N padded 184->192
    float v = 0.f;
    if (n < 184) {
        int deg = n / PPB + 1;
        int m_hk = k % 7 + 1;
        if (deg > m_hk) v = W3[(t * 184 + n) * HH + k];
    }
    int seg = (n >> 4) * 4 + (k >> 5);
    int kl = k & 31;
    int lane = (n & 15) + ((kl >> 3) & 1) * 16;
    w3b[t * (48 * 512) + (seg * 32 + lane) * 16 + (((kl >> 4) & 1) * 8 + (kl & 7))] = (_Float16)v;
}

// ---------------- main flow kernel: 16 samples/block, 4 waves, WMMA MADE + VALU spline ----------
// A fragment (16x32, 16-bit): lane = (m&15) + ((k&31)>=8 && (k&31)<16 ? 16 : ((k&31)>=24 ? 16:0));
// i.e. ln = m + ((kl>>3)&1)*16 ; half j = ((kl>>4)&1)*8 + (kl&7).

__launch_bounds__(128)
__global__ void flow_kernel(const float* __restrict__ y,
                            const float* __restrict__ phi,
                            const float* __restrict__ xg,
                            const float* __restrict__ b1,
                            const float* __restrict__ b2,
                            const float* __restrict__ b3,
                            const _Float16* __restrict__ w1b,
                            const _Float16* __restrict__ w2b,
                            const _Float16* __restrict__ w3b,
                            float* __restrict__ partials, int N)
{
    __shared__ __align__(32) _Float16 xbufA[3 * 512];  // K=96 input fragments
    __shared__ __align__(32) _Float16 hbufA[4 * 512];  // K=128 hidden1 fragments
    __shared__ __align__(32) _Float16 hbufB[4 * 512];  // K=128 hidden2 fragments
    __shared__ float obuf[16 * 192];                   // MADE output (f32)
    __shared__ float ybuf[16 * 8];
    __shared__ float xcbuf[16 * 8];
    __shared__ float ldbuf[16 * 8];
    __shared__ float tld[16];
    __shared__ float psum[16];

    const int tid  = threadIdx.x;
    const int lane = tid & 31;
    const int wave = tid >> 5;
    const int s    = tid & 15;    // sample within tile
    const int dd   = tid >> 4;    // dim 0..7
    const int base = blockIdx.x * 16;

    // ---- init: y tile, ctx into xbufA (k=8..71), zero pad (k=72..95) ----
    {
        int gs = base + s;
        ybuf[s * 8 + dd]  = (gs < N) ? y[gs * DD + dd] : 0.f;
        xcbuf[s * 8 + dd] = 0.f;
        ldbuf[s * 8 + dd] = 0.f;
        if (tid < 16) { tld[tid] = 0.f; psum[tid] = 0.f; }
#pragma unroll
        for (int r = 0; r < 8; ++r) {
            int lin = tid * 8 + r;            // 0..1023 = 16 samples x 64 ctx
            int cs = lin >> 6, cj = lin & 63;
            int gcs = base + cs;
            float cv = 0.f;
            if (gcs < N) cv = (cj < 32) ? phi[gcs * 32 + cj] : xg[gcs * 32 + (cj - 32)];
            int k = 8 + cj;
            int kl = k & 31;
            int ln = cs + ((kl >> 3) & 1) * 16;
            int jj = ((kl >> 4) & 1) * 8 + (kl & 7);
            xbufA[((k >> 5) * 32 + ln) * 16 + jj] = (_Float16)cv;
        }
#pragma unroll
        for (int r = 0; r < 3; ++r) {
            int lin = tid * 3 + r;            // 0..383 = 16 samples x 24 pad
            int cs = lin / 24; int k = 72 + lin % 24;
            int kl = k & 31;
            int ln = cs + ((kl >> 3) & 1) * 16;
            int jj = ((kl >> 4) & 1) * 8 + (kl & 7);
            xbufA[((k >> 5) * 32 + ln) * 16 + jj] = (_Float16)0.f;
        }
    }
    __syncthreads();

    for (int t = TT - 1; t >= 0; --t) {
        const _Float16* w1t = w1b + t * (24 * 512);
        const _Float16* w2t = w2b + t * (32 * 512);
        const _Float16* w3t = w3b + t * (48 * 512);
        const float* b1t = b1 + t * HH;
        const float* b2t = b2 + t * HH;
        const float* b3t = b3 + t * (DD * PPB);

        // 9 MADE evals: it 0..7 = autoregressive inversion steps, it 8 = forward pass for log-det
        for (int it = 0; it <= DD; ++it) {
            // xc -> xbufA k=0..7 (fragment 0, lanes 0..15, halves 0..7)
            xbufA[s * 16 + dd] = (_Float16)xcbuf[s * 8 + dd];
            __syncthreads();

            // ---- layer 1: [16x96] x [96x128] ----
#pragma unroll
            for (int rep = 0; rep < 2; ++rep) {
                int nt = wave + rep * 4;
                v8f acc = {};
#pragma unroll
                for (int ks = 0; ks < 3; ++ks) {
                    v16h a = *(const v16h*)&xbufA[(ks * 32 + lane) * 16];
                    v16h b = *(const v16h*)&w1t[((nt * 3 + ks) * 32 + lane) * 16];
                    acc = __builtin_amdgcn_wmma_f32_16x16x32_f16(false, a, false, b,
                                                                 (short)0, acc, false, false);
                }
                int n = nt * 16 + (lane & 15);
                float bias = b1t[n];
                int mbase = (lane >> 4) * 8;
                int kl = n & 31;
                int lncol = ((kl >> 3) & 1) * 16;
                int jj = ((kl >> 4) & 1) * 8 + (kl & 7);
                _Float16* hb = &hbufA[(n >> 5) * 512];
#pragma unroll
                for (int r = 0; r < 8; ++r) {
                    float hv = fmaxf(acc[r] + bias, 0.f);
                    hb[(lncol + mbase + r) * 16 + jj] = (_Float16)hv;
                }
            }
            __syncthreads();

            // ---- layer 2: [16x128] x [128x128] ----
#pragma unroll
            for (int rep = 0; rep < 2; ++rep) {
                int nt = wave + rep * 4;
                v8f acc = {};
#pragma unroll
                for (int ks = 0; ks < 4; ++ks) {
                    v16h a = *(const v16h*)&hbufA[(ks * 32 + lane) * 16];
                    v16h b = *(const v16h*)&w2t[((nt * 4 + ks) * 32 + lane) * 16];
                    acc = __builtin_amdgcn_wmma_f32_16x16x32_f16(false, a, false, b,
                                                                 (short)0, acc, false, false);
                }
                int n = nt * 16 + (lane & 15);
                float bias = b2t[n];
                int mbase = (lane >> 4) * 8;
                int kl = n & 31;
                int lncol = ((kl >> 3) & 1) * 16;
                int jj = ((kl >> 4) & 1) * 8 + (kl & 7);
                _Float16* hb = &hbufB[(n >> 5) * 512];
#pragma unroll
                for (int r = 0; r < 8; ++r) {
                    float hv = fmaxf(acc[r] + bias, 0.f);
                    hb[(lncol + mbase + r) * 16 + jj] = (_Float16)hv;
                }
            }
            __syncthreads();

            // ---- layer 3: [16x128] x [128x192] -> obuf f32 ----
#pragma unroll
            for (int rep = 0; rep < 3; ++rep) {
                int nt = wave + rep * 4;
                v8f acc = {};
#pragma unroll
                for (int ks = 0; ks < 4; ++ks) {
                    v16h a = *(const v16h*)&hbufB[(ks * 32 + lane) * 16];
                    v16h b = *(const v16h*)&w3t[((nt * 4 + ks) * 32 + lane) * 16];
                    acc = __builtin_amdgcn_wmma_f32_16x16x32_f16(false, a, false, b,
                                                                 (short)0, acc, false, false);
                }
                int n = nt * 16 + (lane & 15);
                float bias = (n < 184) ? b3t[n] : 0.f;
                int mbase = (lane >> 4) * 8;
#pragma unroll
                for (int r = 0; r < 8; ++r)
                    obuf[(mbase + r) * 192 + n] = acc[r] + bias;
            }
            __syncthreads();

            // ---- rational-quadratic spline, one thread per (sample, dim) ----
            {
                const float* pp = &obuf[s * 192 + dd * PPB];
                float wun[8], hun[8], dun[7];
#pragma unroll
                for (int i2 = 0; i2 < 8; ++i2) wun[i2] = pp[i2];
#pragma unroll
                for (int i2 = 0; i2 < 8; ++i2) hun[i2] = pp[8 + i2];
#pragma unroll
                for (int i2 = 0; i2 < 7; ++i2) dun[i2] = pp[16 + i2];

                float widths[8], heights[8], cwa[9], cha[9], der[9];
                const float scale = (1.f - MINBIN * (float)KKB) * (2.f * BOUNDF);
                const float floorw = MINBIN * 2.f * BOUNDF;

                float mx = wun[0];
#pragma unroll
                for (int i2 = 1; i2 < 8; ++i2) mx = fmaxf(mx, wun[i2]);
                float se = 0.f;
#pragma unroll
                for (int i2 = 0; i2 < 8; ++i2) { widths[i2] = expf(wun[i2] - mx); se += widths[i2]; }
                float isw = 1.f / se;
#pragma unroll
                for (int i2 = 0; i2 < 8; ++i2) widths[i2] = floorw + scale * widths[i2] * isw;

                mx = hun[0];
#pragma unroll
                for (int i2 = 1; i2 < 8; ++i2) mx = fmaxf(mx, hun[i2]);
                se = 0.f;
#pragma unroll
                for (int i2 = 0; i2 < 8; ++i2) { heights[i2] = expf(hun[i2] - mx); se += heights[i2]; }
                float ish = 1.f / se;
#pragma unroll
                for (int i2 = 0; i2 < 8; ++i2) heights[i2] = floorw + scale * heights[i2] * ish;

                cwa[0] = -BOUNDF; cha[0] = -BOUNDF;
#pragma unroll
                for (int i2 = 0; i2 < 8; ++i2) {
                    cwa[i2 + 1] = cwa[i2] + widths[i2];
                    cha[i2 + 1] = cha[i2] + heights[i2];
                }
                der[0] = 1.f; der[8] = 1.f;
#pragma unroll
                for (int i2 = 0; i2 < 7; ++i2) {
                    float xx = dun[i2];
                    der[i2 + 1] = MINBIN + fmaxf(xx, 0.f) + log1pf(expf(-fabsf(xx)));
                }

                bool inverse = (it < DD);
                float val = inverse ? ybuf[s * 8 + dd] : xcbuf[s * 8 + dd];
                bool inside = (val > -BOUNDF) && (val < BOUNDF);
                float v = fminf(fmaxf(val, -BOUNDF + 1e-6f), BOUNDF - 1e-6f);

                int idx = 0;
#pragma unroll
                for (int i2 = 1; i2 <= 7; ++i2) {
                    float kn = inverse ? cha[i2] : cwa[i2];
                    idx += (kn <= v) ? 1 : 0;
                }
                float xk = cwa[0], yk = cha[0], wk = widths[0], hk = heights[0];
                float dk = der[0], dk1 = der[1];
#pragma unroll
                for (int i2 = 1; i2 < 8; ++i2) {
                    bool m = (idx == i2);
                    xk  = m ? cwa[i2]     : xk;   yk  = m ? cha[i2]     : yk;
                    wk  = m ? widths[i2]  : wk;   hk  = m ? heights[i2] : hk;
                    dk  = m ? der[i2]     : dk;   dk1 = m ? der[i2 + 1] : dk1;
                }
                float srq = hk / wk;
                float m2s = dk1 + dk - 2.f * srq;
                if (inverse) {
                    float tq = v - yk;
                    float aq = hk * (srq - dk) + tq * m2s;
                    float bq = hk * dk - tq * m2s;
                    float cq = -srq * tq;
                    float disc = fmaxf(bq * bq - 4.f * aq * cq, 0.f);
                    float xi = 2.f * cq / (-bq - sqrtf(disc));
                    float out = xi * wk + xk;
                    xcbuf[s * 8 + dd] = inside ? out : val;
                } else {
                    float xi = (v - xk) / wk;
                    float omxi = 1.f - xi;
                    float den = srq + m2s * xi * omxi;
                    float num = dk1 * xi * xi + 2.f * srq * xi * omxi + dk * omxi * omxi;
                    float ld = 2.f * logf(srq) + logf(num) - 2.f * logf(den);
                    ldbuf[s * 8 + dd] = inside ? ld : 0.f;
                }
            }
            __syncthreads();
        } // it

        // accumulate log-det, roll state: xcur <- xinv, xc <- 0
        if (tid < 16) {
            float a2 = 0.f;
#pragma unroll
            for (int d2 = 0; d2 < 8; ++d2) a2 += ldbuf[tid * 8 + d2];
            tld[tid] += a2;
        }
        float newy = xcbuf[s * 8 + dd];
        __syncthreads();
        ybuf[s * 8 + dd]  = newy;
        xcbuf[s * 8 + dd] = 0.f;
        __syncthreads();
    } // t

    // base log-prob and per-block partial of (-lp), fixed-order sum
    if (tid < 16) {
        float a2 = 0.f;
#pragma unroll
        for (int d2 = 0; d2 < 8; ++d2) {
            float xv = ybuf[tid * 8 + d2];
            a2 += -0.5f * xv * xv - 0.91893853320467274f;
        }
        psum[tid] = -(a2 - tld[tid]);
    }
    __syncthreads();
    if (tid == 0) {
        int lim = N - base; if (lim > 16) lim = 16;
        float tot = 0.f;
        for (int i2 = 0; i2 < 16; ++i2) if (i2 < lim) tot += psum[i2];
        partials[blockIdx.x] = tot;
    }
}

// ---------------- deterministic final reduction ----------------
__global__ void reduce_kernel(const float* __restrict__ partials, int nPart,
                              const float* __restrict__ y_std, int N,
                              float* __restrict__ out) {
    __shared__ float buf[256];
    int tid = threadIdx.x;
    float a = 0.f;
    for (int i = tid; i < nPart; i += 256) a += partials[i];
    buf[tid] = a;
    __syncthreads();
    for (int s2 = 128; s2 > 0; s2 >>= 1) {
        if (tid < s2) buf[tid] += buf[tid + s2];
        __syncthreads();
    }
    if (tid == 0) {
        float sls = 0.f;
        for (int d = 0; d < DD; ++d) sls += logf(y_std[d]);
        out[0] = buf[0] / (float)N + sls;
    }
}

extern "C" void kernel_launch(void* const* d_in, const int* in_sizes, int n_in,
                              void* d_out, int out_size, void* d_ws, size_t ws_size,
                              hipStream_t stream) {
    const float* y     = (const float*)d_in[0];
    const float* phi   = (const float*)d_in[1];
    const float* x     = (const float*)d_in[2];
    const float* y_std = (const float*)d_in[3];
    const float* W1    = (const float*)d_in[4];
    const float* b1    = (const float*)d_in[5];
    const float* W2    = (const float*)d_in[6];
    const float* b2    = (const float*)d_in[7];
    const float* W3    = (const float*)d_in[8];
    const float* b3    = (const float*)d_in[9];

    int N = in_sizes[0] / DD;

    // workspace layout (all 32B aligned): swizzled f16 masked weights + per-block partials
    char* ws = (char*)d_ws;
    _Float16* w1b = (_Float16*)(ws);                    // 5*24*512 halves = 122880 B
    _Float16* w2b = (_Float16*)(ws + 122880);           // 5*32*512 halves = 163840 B
    _Float16* w3b = (_Float16*)(ws + 286720);           // 5*48*512 halves = 245760 B
    float* partials = (float*)(ws + 532480);            // nBlocks floats

    prep_w1<<<(TT * 96 * 128 + 255) / 256, 256, 0, stream>>>(W1, w1b);
    prep_w2<<<(TT * 128 * 128 + 255) / 256, 256, 0, stream>>>(W2, w2b);
    prep_w3<<<(TT * 128 * 192 + 255) / 256, 256, 0, stream>>>(W3, w3b);

    int nBlocks = (N + 15) / 16;
    flow_kernel<<<nBlocks, 128, 0, stream>>>(y, phi, x, b1, b2, b3,
                                             w1b, w2b, w3b, partials, N);
    reduce_kernel<<<1, 256, 0, stream>>>(partials, nBlocks, y_std, N, (float*)d_out);
}
